// DQN_GNN_52630529245833
// MI455X (gfx1250) — compile-verified, matrix operand order checked by reading, not run on hardware
//
#include <hip/hip_runtime.h>
#include <hip/hip_bf16.h>

// ---------------------------------------------------------------------------
// Types for CDNA5 WMMA (gfx1250, wave32)
// ---------------------------------------------------------------------------
typedef __attribute__((ext_vector_type(16))) __bf16 v16bf;
typedef __attribute__((ext_vector_type(8)))  __bf16 v8bf;
typedef __attribute__((ext_vector_type(8)))  float  v8f;

#define D_H 128   // feature width (both layers)
#define N_ACT 16  // FC output width

// ---------------------------------------------------------------------------
// Fragment loader: 16-bit A/B matrix 16x32 lane layout (ISA 7.12.2).
// lane in [0,16): K chunks [kbase+0..7] then [kbase+16..23]
// lane in [16,32): K chunks [kbase+8..15] then [kbase+24..31]
// Each chunk is 8 contiguous bf16 = one b128 load.
// ---------------------------------------------------------------------------
__device__ __forceinline__ v16bf load_frag_row(const __bf16* __restrict__ row,
                                               int kbase, int kg) {
    const v8bf lo = *reinterpret_cast<const v8bf*>(row + kbase + 8 * kg);
    const v8bf hi = *reinterpret_cast<const v8bf*>(row + kbase + 16 + 8 * kg);
    v16bf r;
#pragma unroll
    for (int i = 0; i < 8; ++i) { r[i] = lo[i]; r[8 + i] = hi[i]; }
    return r;
}

// ---------------------------------------------------------------------------
// GEMM: C[M x 128] = A[M x 128] * W[128 x 128], A bf16 row-major,
// Bt bf16 = W transposed (Bt[n][k] = W[k][n]) so B frags are contiguous.
// One wave -> one 16x16 tile; 8 waves/block -> full 128 columns of a 16-row
// stripe. grid.x = M/16.
// ---------------------------------------------------------------------------
__global__ void gemm_bf16_128(const __bf16* __restrict__ A,
                              const __bf16* __restrict__ Bt,
                              float* __restrict__ C) {
    const int lane = threadIdx.x & 31;
    const int wave = threadIdx.x >> 5;
    const int row0 = blockIdx.x * 16;
    const int col0 = wave * 16;
    const int m  = lane & 15;
    const int kg = lane >> 4;

    const __bf16* arow = A  + (size_t)(row0 + m) * D_H;
    const __bf16* brow = Bt + (size_t)(col0 + m) * D_H;

    v8f acc = {};
#pragma unroll
    for (int kk = 0; kk < 4; ++kk) {
        v16bf a = load_frag_row(arow, kk * 32, kg);
        v16bf b = load_frag_row(brow, kk * 32, kg);
        acc = __builtin_amdgcn_wmma_f32_16x16x32_bf16(
            /*neg_a=*/false, a, /*neg_b=*/false, b,
            /*c_mod=*/(short)0, acc, /*reuse_a=*/false, /*reuse_b=*/false);
    }

    // C/D layout: VGPR r holds M=r (lanes 0-15) and M=r+8 (lanes 16-31), N=lane&15
    const int n = col0 + (lane & 15);
    const int rbase = row0 + 8 * kg;
#pragma unroll
    for (int r = 0; r < 8; ++r)
        C[(size_t)(rbase + r) * D_H + n] = acc[r];
}

// ---------------------------------------------------------------------------
// FC head: out[M x 16] = H[M x 128] * Wfc[128 x 16] + bfc, WMMA + fused bias.
// One wave per 16-row tile (N=16 exactly). 8 waves/block.
// ---------------------------------------------------------------------------
__global__ void fc_wmma(const __bf16* __restrict__ A,
                        const __bf16* __restrict__ Bt,   // [16][128]
                        const float* __restrict__ bias,  // [16]
                        float* __restrict__ out, int mtiles) {
    const int lane = threadIdx.x & 31;
    const int wave = threadIdx.x >> 5;
    const int tile = blockIdx.x * 8 + wave;
    if (tile >= mtiles) return;  // wave-uniform exit; EXEC stays all-1s for WMMA
    const int row0 = tile * 16;
    const int m  = lane & 15;
    const int kg = lane >> 4;

    const __bf16* arow = A  + (size_t)(row0 + m) * D_H;
    const __bf16* brow = Bt + (size_t)m * D_H;

    v8f acc = {};
#pragma unroll
    for (int kk = 0; kk < 4; ++kk) {
        v16bf a = load_frag_row(arow, kk * 32, kg);
        v16bf b = load_frag_row(brow, kk * 32, kg);
        acc = __builtin_amdgcn_wmma_f32_16x16x32_bf16(
            false, a, false, b, (short)0, acc, false, false);
    }

    const int n = lane & 15;
    const float bn = bias[n];
    const int rbase = row0 + 8 * kg;
#pragma unroll
    for (int r = 0; r < 8; ++r)
        out[(size_t)(rbase + r) * N_ACT + n] = acc[r] + bn;
}

// ---------------------------------------------------------------------------
// Degree / normalization helpers
// ---------------------------------------------------------------------------
__global__ void deg_init(float* __restrict__ deg, int n) {
    int i = blockIdx.x * blockDim.x + threadIdx.x;
    if (i < n) deg[i] = 1.0f;  // self-loop contribution
}

__global__ void deg_edges(const int* __restrict__ dst, float* __restrict__ deg, int e) {
    int i = blockIdx.x * blockDim.x + threadIdx.x;
    if (i < e) atomicAdd(&deg[dst[i]], 1.0f);
}

__global__ void deg_rsqrt(float* __restrict__ dinv, int n) {
    int i = blockIdx.x * blockDim.x + threadIdx.x;
    if (i < n) dinv[i] = rsqrtf(dinv[i]);  // deg >= 1 always
}

// ---------------------------------------------------------------------------
// Elementwise: cast f32 -> bf16
// ---------------------------------------------------------------------------
__global__ void cast_bf16(const float* __restrict__ src, __bf16* __restrict__ dst,
                          size_t total) {
    size_t i = (size_t)blockIdx.x * blockDim.x + threadIdx.x;
    if (i < total) dst[i] = (__bf16)src[i];
}

// Transpose + cast: dst[n*K + k] = src[k*N + n]   (weights -> column-major bf16)
__global__ void transpose_cast(const float* __restrict__ src, __bf16* __restrict__ dst,
                               int K, int Ncols, int total) {
    int i = blockIdx.x * blockDim.x + threadIdx.x;
    if (i >= total) return;
    int n = i / K, k = i % K;
    dst[i] = (__bf16)src[(size_t)k * Ncols + n];
}

// ---------------------------------------------------------------------------
// Aggregation: agg[i] = xw[i] * dinv[i]^2 (self loop; also initializes agg)
// ---------------------------------------------------------------------------
__global__ void selfloop_init(const float* __restrict__ xw,
                              const float* __restrict__ dinv,
                              float* __restrict__ agg, int n) {
    int idx = blockIdx.x * blockDim.x + threadIdx.x;
    int i = idx >> 5;
    if (i >= n) return;
    int fg = (idx & 31) * 4;
    float nm = dinv[i] * dinv[i];
    const float4 v = *reinterpret_cast<const float4*>(xw + (size_t)i * D_H + fg);
    float4 o = make_float4(v.x * nm, v.y * nm, v.z * nm, v.w * nm);
    *reinterpret_cast<float4*>(agg + (size_t)i * D_H + fg) = o;
}

// Edge scatter: agg[dst] += xw[src] * dinv[src]*dinv[dst]   (float4 + 4 atomics)
__global__ void edge_scatter(const int* __restrict__ src, const int* __restrict__ dst,
                             const float* __restrict__ dinv,
                             const float* __restrict__ xw,
                             float* __restrict__ agg, int e) {
    int idx = blockIdx.x * blockDim.x + threadIdx.x;
    int ei = idx >> 5;
    if (ei >= e) return;
    int fg = (idx & 31) * 4;
    int s = src[ei], d = dst[ei];
    float nm = dinv[s] * dinv[d];
    const float4 v = *reinterpret_cast<const float4*>(xw + (size_t)s * D_H + fg);
    float* o = agg + (size_t)d * D_H + fg;
    atomicAdd(o + 0, v.x * nm);
    atomicAdd(o + 1, v.y * nm);
    atomicAdd(o + 2, v.z * nm);
    atomicAdd(o + 3, v.w * nm);
}

// h = relu(agg + b) -> bf16 (feeds the next WMMA GEMM)
__global__ void bias_relu_cast(const float* __restrict__ agg,
                               const float* __restrict__ b,
                               __bf16* __restrict__ hb, size_t total) {
    size_t i = (size_t)blockIdx.x * blockDim.x + threadIdx.x;
    if (i >= total) return;
    int f = (int)(i & (D_H - 1));
    float h = agg[i] + b[f];
    hb[i] = (__bf16)fmaxf(h, 0.0f);
}

// ---------------------------------------------------------------------------
// Launcher
// ---------------------------------------------------------------------------
extern "C" void kernel_launch(void* const* d_in, const int* in_sizes, int n_in,
                              void* d_out, int out_size, void* d_ws, size_t ws_size,
                              hipStream_t stream) {
    const float* x    = (const float*)d_in[0];
    const int*   edge = (const int*)d_in[1];   // [2][E]: row0=src, row1=dst
    const float* W1   = (const float*)d_in[2];
    const float* b1   = (const float*)d_in[3];
    const float* W2   = (const float*)d_in[4];
    const float* b2   = (const float*)d_in[5];
    const float* Wfc  = (const float*)d_in[6];
    const float* bfc  = (const float*)d_in[7];
    float* out = (float*)d_out;

    const int N = in_sizes[0] / D_H;   // 50000
    const int E = in_sizes[1] / 2;     // 800000
    const int* esrc = edge;
    const int* edst = edge + E;

    // Workspace carve-out (256B-aligned), fits in L2 (~65 MB total)
    char* ws = (char*)d_ws;
    size_t off = 0;
    auto carve = [&](size_t bytes) {
        char* p = ws + off;
        off = (off + bytes + 255) & ~(size_t)255;
        return p;
    };
    float*  dinv = (float*)carve((size_t)N * sizeof(float));
    __bf16* Xb   = (__bf16*)carve((size_t)N * D_H * sizeof(__bf16)); // reused for h1b/h2b
    __bf16* W1t  = (__bf16*)carve((size_t)D_H * D_H * sizeof(__bf16));
    __bf16* W2t  = (__bf16*)carve((size_t)D_H * D_H * sizeof(__bf16));
    __bf16* Wfct = (__bf16*)carve((size_t)N_ACT * D_H * sizeof(__bf16));
    float*  xw   = (float*)carve((size_t)N * D_H * sizeof(float));
    float*  agg  = (float*)carve((size_t)N * D_H * sizeof(float));
    if (off > ws_size) return;  // insufficient scratch (shouldn't happen)

    const int TB = 256;
    const size_t NF = (size_t)N * D_H;
    const int mtiles = N / 16;                  // 3125
    dim3 gemm_grid(mtiles), gemm_blk(256);      // 8 waves/block = 128 cols

    // 1) degree -> dinv
    deg_init<<<(N + TB - 1) / TB, TB, 0, stream>>>(dinv, N);
    deg_edges<<<(E + TB - 1) / TB, TB, 0, stream>>>(edst, dinv, E);
    deg_rsqrt<<<(N + TB - 1) / TB, TB, 0, stream>>>(dinv, N);

    // 2) precision packing: X -> bf16, weights -> transposed bf16
    cast_bf16<<<(int)((NF + TB - 1) / TB), TB, 0, stream>>>(x, Xb, NF);
    transpose_cast<<<(D_H * D_H + TB - 1) / TB, TB, 0, stream>>>(W1, W1t, D_H, D_H, D_H * D_H);
    transpose_cast<<<(D_H * D_H + TB - 1) / TB, TB, 0, stream>>>(W2, W2t, D_H, D_H, D_H * D_H);
    transpose_cast<<<(N_ACT * D_H + TB - 1) / TB, TB, 0, stream>>>(Wfc, Wfct, D_H, N_ACT, N_ACT * D_H);

    // ---- layer 1: xw = X W1 ; agg = normalized scatter ; h1 = relu(agg+b1)
    gemm_bf16_128<<<gemm_grid, gemm_blk, 0, stream>>>(Xb, W1t, xw);
    selfloop_init<<<(N * 32 + TB - 1) / TB, TB, 0, stream>>>(xw, dinv, agg, N);
    edge_scatter<<<(int)(((size_t)E * 32 + TB - 1) / TB), TB, 0, stream>>>(esrc, edst, dinv, xw, agg, E);
    bias_relu_cast<<<(int)((NF + TB - 1) / TB), TB, 0, stream>>>(agg, b1, Xb, NF);

    // ---- layer 2
    gemm_bf16_128<<<gemm_grid, gemm_blk, 0, stream>>>(Xb, W2t, xw);
    selfloop_init<<<(N * 32 + TB - 1) / TB, TB, 0, stream>>>(xw, dinv, agg, N);
    edge_scatter<<<(int)(((size_t)E * 32 + TB - 1) / TB), TB, 0, stream>>>(esrc, edst, dinv, xw, agg, E);
    bias_relu_cast<<<(int)((NF + TB - 1) / TB), TB, 0, stream>>>(agg, b2, Xb, NF);

    // ---- FC head: out = h2 Wfc + bfc
    fc_wmma<<<(mtiles + 7) / 8, 256, 0, stream>>>(Xb, Wfct, bfc, out, mtiles);
}
